// SelfAttention_78975858638873
// MI455X (gfx1250) — matrix operand order repaired and checked by reading, not compiled, not run on hardware
//
#include <hip/hip_runtime.h>
#include <math.h>

#define BB   8
#define CC   256
#define NN   4096
#define DQK  64
#define DOUT 128

typedef __attribute__((ext_vector_type(16))) __bf16 v16bf;
typedef __attribute__((ext_vector_type(8)))  __bf16 v8bf;
typedef __attribute__((ext_vector_type(8)))  float  v8f;

static __device__ __forceinline__ __bf16 f2bf(float f) {
  unsigned u = __builtin_bit_cast(unsigned, f);
  u += 0x7FFFu + ((u >> 16) & 1u);          // round-to-nearest-even
  unsigned short h = (unsigned short)(u >> 16);
  return __builtin_bit_cast(__bf16, h);
}

// 16-lane-group reductions via DPP16 (no LDS, no DS waits):
// xor1 (quad_perm), xor2 (quad_perm), row_half_mirror, row_mirror.
static __device__ __forceinline__ float red_max16(float x) {
  int t; float y;
  t = __builtin_amdgcn_update_dpp(0, __builtin_bit_cast(int, x), 0xB1,  0xF, 0xF, true);
  y = __builtin_bit_cast(float, t); x = fmaxf(x, y);
  t = __builtin_amdgcn_update_dpp(0, __builtin_bit_cast(int, x), 0x4E,  0xF, 0xF, true);
  y = __builtin_bit_cast(float, t); x = fmaxf(x, y);
  t = __builtin_amdgcn_update_dpp(0, __builtin_bit_cast(int, x), 0x141, 0xF, 0xF, true);
  y = __builtin_bit_cast(float, t); x = fmaxf(x, y);
  t = __builtin_amdgcn_update_dpp(0, __builtin_bit_cast(int, x), 0x140, 0xF, 0xF, true);
  y = __builtin_bit_cast(float, t); x = fmaxf(x, y);
  return x;
}
static __device__ __forceinline__ float red_sum16(float x) {
  int t;
  t = __builtin_amdgcn_update_dpp(0, __builtin_bit_cast(int, x), 0xB1,  0xF, 0xF, true);
  x += __builtin_bit_cast(float, t);
  t = __builtin_amdgcn_update_dpp(0, __builtin_bit_cast(int, x), 0x4E,  0xF, 0xF, true);
  x += __builtin_bit_cast(float, t);
  t = __builtin_amdgcn_update_dpp(0, __builtin_bit_cast(int, x), 0x141, 0xF, 0xF, true);
  x += __builtin_bit_cast(float, t);
  t = __builtin_amdgcn_update_dpp(0, __builtin_bit_cast(int, x), 0x140, 0xF, 0xF, true);
  x += __builtin_bit_cast(float, t);
  return x;
}

static __device__ __forceinline__ v16bf make_a(const __bf16* r, int koff) {
  v8bf lo = *(const v8bf*)(r + 2 * koff);
  v8bf hi = *(const v8bf*)(r + 2 * koff + 8);
  v16bf a;
#pragma unroll
  for (int i = 0; i < 8; ++i) { a[i] = lo[i]; a[8 + i] = hi[i]; }
  return a;
}

// ---------------------------------------------------------------------------
// Kernel 0a: f32 -> bf16 copy (weights)
// ---------------------------------------------------------------------------
__global__ __launch_bounds__(256) void cvt_kernel(const float* __restrict__ s,
                                                  __bf16* __restrict__ d, int n) {
  int i = blockIdx.x * 256 + threadIdx.x;
  if (i < n) d[i] = f2bf(s[i]);
}

// ---------------------------------------------------------------------------
// Kernel 0b: x [b][c][n] f32 -> xbf [b][n][c] bf16 (channel-contiguous so
// WMMA B-operands are single 32B/lane loads).  32x32 LDS tile transpose.
// ---------------------------------------------------------------------------
__global__ __launch_bounds__(256) void xpose_kernel(const float* __restrict__ x,
                                                    __bf16* __restrict__ xbf) {
  __shared__ float t[32][33];
  int n0 = blockIdx.x << 5, c0 = blockIdx.y << 5, b = blockIdx.z;
  int i = threadIdx.x >> 5, j = threadIdx.x & 31;
#pragma unroll
  for (int r = 0; r < 4; ++r) {
    int c = c0 + i + (r << 3);
    t[i + (r << 3)][j] = x[((size_t)b * CC + c) * NN + n0 + j];
  }
  __syncthreads();
#pragma unroll
  for (int r = 0; r < 4; ++r) {
    int n = n0 + i + (r << 3);
    xbf[((size_t)b * NN + n) * CC + c0 + j] = f2bf(t[j][i + (r << 3)]);
  }
}

// ---------------------------------------------------------------------------
// Kernel 1: QKV projection as WMMA GEMM.  1 wave per 16-pixel tile; gy selects
// output group (0: q+k = 8 row-tiles, 1: v[0:128), 2: v[128:256)).
// Q,K written [b][n][64]; V written [b][c][n] (attention operand layouts).
// ---------------------------------------------------------------------------
__global__ __launch_bounds__(32) void qkv_wmma_kernel(
    const __bf16* __restrict__ xbf,
    const __bf16* __restrict__ wqbf, const __bf16* __restrict__ wkbf,
    const __bf16* __restrict__ wvbf,
    const float* __restrict__ bq, const float* __restrict__ bk,
    const float* __restrict__ bv,
    __bf16* __restrict__ qws, __bf16* __restrict__ kws, __bf16* __restrict__ vws)
{
  int g = blockIdx.y, b = blockIdx.z, n0 = blockIdx.x << 4;
  int l = threadIdx.x;
  int lhalf = l >> 4;
  int m     = l & 15;
  int koff  = lhalf ? 8  : 0;
  int kb16  = lhalf ? 16 : 0;

  const __bf16* xb = xbf + ((size_t)b * NN + n0 + m) * CC;

  const __bf16* wrow[8];
#pragma unroll
  for (int t = 0; t < 8; ++t) {
    if (g == 0) wrow[t] = (t < 4) ? wqbf + (size_t)(16 * t + m) * CC
                                  : wkbf + (size_t)(16 * (t - 4) + m) * CC;
    else        wrow[t] = wvbf + (size_t)(((g == 1) ? 0 : 128) + 16 * t + m) * CC;
  }

  v8f zacc = {};
  v8f acc[8];
#pragma unroll
  for (int t = 0; t < 8; ++t) acc[t] = zacc;

  for (int c0 = 0; c0 < CC; c0 += 32) {
    v16bf xop = *(const v16bf*)(xb + c0 + kb16);
#pragma unroll
    for (int t = 0; t < 8; ++t) {
      v16bf a = make_a(wrow[t] + c0, koff);
      acc[t] = __builtin_amdgcn_wmma_f32_16x16x32_bf16(false, a, false, xop, (short)0, acc[t], false, false);
    }
  }

  if (g == 0) {
#pragma unroll
    for (int t = 0; t < 8; ++t) {
      int dbase = 16 * (t & 3) + (lhalf ? 8 : 0);
      const float* bias = (t < 4) ? bq : bk;
      v8bf outv;
#pragma unroll
      for (int v = 0; v < 8; ++v) outv[v] = f2bf(acc[t][v] + bias[dbase + v]);
      __bf16* dst = ((t < 4) ? qws : kws) + ((size_t)b * NN + n0 + m) * DQK + dbase;
      *(v8bf*)dst = outv;
    }
  } else {
    int rb = (g == 1) ? 0 : 128;
#pragma unroll
    for (int t = 0; t < 8; ++t)
#pragma unroll
      for (int v = 0; v < 8; ++v) {
        int d = rb + 16 * t + (lhalf ? 8 : 0) + v;
        vws[((size_t)b * CC + d) * NN + n0 + m] = f2bf(acc[t][v] + bv[d]);
      }
  }
}

// ---------------------------------------------------------------------------
// Kernel 2: flash attention, 1 wave per 16-query tile.  S = Q K^T (4 WMMAs,
// K-dim 64 chained), online softmax with DPP row reductions, O += P V with
// 16 accumulator tiles.  Next K B-operands prefetched behind the S-WMMAs;
// V B-operands issued in clause groups of 4.  Epilogue fuses gamma*O/l + x
// and writes bf16 out^T [b][n][256] for the MLP.
// ---------------------------------------------------------------------------
__global__ __launch_bounds__(32) void attn_kernel(
    const __bf16* __restrict__ qws, const __bf16* __restrict__ kws,
    const __bf16* __restrict__ vws, const float* __restrict__ x,
    const float* __restrict__ gamma, __bf16* __restrict__ outT)
{
  __shared__ float sP[16 * 32];
  int b  = blockIdx.y;
  int q0 = blockIdx.x << 4;
  int l  = threadIdx.x;
  int lhalf = l >> 4;
  int m     = l & 15;
  int koff  = lhalf ? 8  : 0;
  int kb16  = lhalf ? 16 : 0;

  // Q tile (16 x 64) as two bf16 A-operands in registers
  const __bf16* qp = qws + ((size_t)b * NN + q0 + m) * DQK;
  v16bf qa0 = make_a(qp, koff);
  v16bf qa1 = make_a(qp + 32, koff);

  v8f zacc = {};
  v8f o[16];
#pragma unroll
  for (int cb = 0; cb < 16; ++cb) o[cb] = zacc;
  float mrow[8], lsum[8];
#pragma unroll
  for (int v = 0; v < 8; ++v) { mrow[v] = -3.0e38f; lsum[v] = 0.0f; }

  const __bf16* kp = kws + (size_t)b * NN * DQK;
  const __bf16* vp = vws + (size_t)b * CC * NN;

#define LOADK(dst, jb)                                                        \
  do {                                                                        \
    dst[0] = *(const v16bf*)(kp + (size_t)((jb) + m) * DQK + kb16);           \
    dst[1] = *(const v16bf*)(kp + (size_t)((jb) + m) * DQK + 32 + kb16);      \
    dst[2] = *(const v16bf*)(kp + (size_t)((jb) + 16 + m) * DQK + kb16);      \
    dst[3] = *(const v16bf*)(kp + (size_t)((jb) + 16 + m) * DQK + 32 + kb16); \
  } while (0)

  v16bf kb[4];
  LOADK(kb, 0);

  for (int j0 = 0; j0 < NN; j0 += 32) {
    v8f s0 = __builtin_amdgcn_wmma_f32_16x16x32_bf16(false, qa0, false, kb[0], (short)0, zacc, false, false);
    s0     = __builtin_amdgcn_wmma_f32_16x16x32_bf16(false, qa1, false, kb[1], (short)0, s0,   false, false);
    v8f s1 = __builtin_amdgcn_wmma_f32_16x16x32_bf16(false, qa0, false, kb[2], (short)0, zacc, false, false);
    s1     = __builtin_amdgcn_wmma_f32_16x16x32_bf16(false, qa1, false, kb[3], (short)0, s1,   false, false);

    // prefetch next iteration's K operands (wraps harmlessly on last iter)
    int jn = j0 + 32;
    if (jn >= NN) jn = 0;
    LOADK(kb, jn);

    // online softmax; stats share the accumulator (half,vgpr)->row layout
    float alpha[8];
#pragma unroll
    for (int v = 0; v < 8; ++v) {
      float t  = red_max16(fmaxf(s0[v], s1[v]));
      float mn = fmaxf(mrow[v], t);
      alpha[v] = __expf(mrow[v] - mn);
      mrow[v]  = mn;
      float p0 = __expf(s0[v] - mn);
      float p1 = __expf(s1[v] - mn);
      s0[v] = p0; s1[v] = p1;
      lsum[v] = lsum[v] * alpha[v] + red_sum16(p0 + p1);
    }
#pragma unroll
    for (int cb = 0; cb < 16; ++cb)
#pragma unroll
      for (int v = 0; v < 8; ++v) o[cb][v] *= alpha[v];

    // transpose P (C-layout) -> bf16 A-layout via LDS
    __syncthreads();
#pragma unroll
    for (int v = 0; v < 8; ++v) {
      int row = (lhalf ? 8 : 0) + v;
      sP[row * 32 + m]      = s0[v];
      sP[row * 32 + 16 + m] = s1[v];
    }
    __syncthreads();
    v16bf pa;
#pragma unroll
    for (int e = 0; e < 8; ++e)  pa[e] = f2bf(sP[m * 32 + 2 * koff + e]);
#pragma unroll
    for (int e = 8; e < 16; ++e) pa[e] = f2bf(sP[m * 32 + 2 * koff + e]);

    // O[16x256] += P[16x32] * V[32x256] : 16 WMMAs, V loads in groups of 4
#pragma unroll
    for (int gq = 0; gq < 4; ++gq) {
      v16bf vb0 = *(const v16bf*)(vp + (size_t)((gq * 4 + 0) * 16 + m) * NN + j0 + kb16);
      v16bf vb1 = *(const v16bf*)(vp + (size_t)((gq * 4 + 1) * 16 + m) * NN + j0 + kb16);
      v16bf vb2 = *(const v16bf*)(vp + (size_t)((gq * 4 + 2) * 16 + m) * NN + j0 + kb16);
      v16bf vb3 = *(const v16bf*)(vp + (size_t)((gq * 4 + 3) * 16 + m) * NN + j0 + kb16);
      o[gq * 4 + 0] = __builtin_amdgcn_wmma_f32_16x16x32_bf16(false, pa, false, vb0, (short)0, o[gq * 4 + 0], false, false);
      o[gq * 4 + 1] = __builtin_amdgcn_wmma_f32_16x16x32_bf16(false, pa, false, vb1, (short)0, o[gq * 4 + 1], false, false);
      o[gq * 4 + 2] = __builtin_amdgcn_wmma_f32_16x16x32_bf16(false, pa, false, vb2, (short)0, o[gq * 4 + 2], false, false);
      o[gq * 4 + 3] = __builtin_amdgcn_wmma_f32_16x16x32_bf16(false, pa, false, vb3, (short)0, o[gq * 4 + 3], false, false);
    }
    __syncthreads();
  }
#undef LOADK

  // epilogue: outT[n][ch] = bf16( gamma * O/l + x )
  float g = gamma[0];
  float inv[8];
#pragma unroll
  for (int v = 0; v < 8; ++v) inv[v] = 1.0f / lsum[v];

  const float* xb = x    + (size_t)b * CC * NN;
  __bf16*      ot = outT + (size_t)b * NN * CC;
#pragma unroll
  for (int cb = 0; cb < 16; ++cb) {
    int ch = cb * 16 + m;
#pragma unroll
    for (int v = 0; v < 8; ++v) {
      int n = q0 + (lhalf ? 8 : 0) + v;
      float val = g * (o[cb][v] * inv[v]) + xb[(size_t)ch * NN + n];
      ot[(size_t)n * CC + ch] = f2bf(val);
    }
  }
}

// ---------------------------------------------------------------------------
// Kernel 3: fused MLP as WMMA GEMMs.  1 wave per 16-pixel tile.
// hid = relu(w1 @ out + b1) staged through LDS (transpose to B-layout),
// y = w2 @ hid + b2 -> f32 d_out.
// ---------------------------------------------------------------------------
__global__ __launch_bounds__(32) void mlp_wmma_kernel(
    const __bf16* __restrict__ outT, const __bf16* __restrict__ w1bf,
    const float* __restrict__ b1,    const __bf16* __restrict__ w2bf,
    const float* __restrict__ b2,    float* __restrict__ y)
{
  __shared__ __bf16 hT[16 * 144];    // [pixel][hid], stride 144 keeps 32B align
  int b = blockIdx.y, n0 = blockIdx.x << 4;
  int l = threadIdx.x;
  int lhalf = l >> 4;
  int m     = l & 15;
  int koff  = lhalf ? 8  : 0;
  int kb16  = lhalf ? 16 : 0;

  const __bf16* ob = outT + ((size_t)b * NN + n0 + m) * CC;

  v8f zacc = {};
  v8f acc[8];
#pragma unroll
  for (int t = 0; t < 8; ++t) acc[t] = zacc;

  for (int c0 = 0; c0 < CC; c0 += 32) {
    v16bf xop = *(const v16bf*)(ob + c0 + kb16);
#pragma unroll
    for (int t = 0; t < 8; ++t) {
      v16bf a = make_a(w1bf + (size_t)(16 * t + m) * CC + c0, koff);
      acc[t] = __builtin_amdgcn_wmma_f32_16x16x32_bf16(false, a, false, xop, (short)0, acc[t], false, false);
    }
  }
#pragma unroll
  for (int t = 0; t < 8; ++t)
#pragma unroll
    for (int v = 0; v < 8; ++v) {
      int d = 16 * t + (lhalf ? 8 : 0) + v;
      hT[m * 144 + d] = f2bf(fmaxf(acc[t][v] + b1[d], 0.0f));
    }
  __syncthreads();

  v8f acc2[8];
#pragma unroll
  for (int t = 0; t < 8; ++t) acc2[t] = zacc;

  for (int c0 = 0; c0 < DOUT; c0 += 32) {
    v16bf hop = *(const v16bf*)(&hT[m * 144 + c0 + kb16]);
#pragma unroll
    for (int t = 0; t < 8; ++t) {
      v16bf a = make_a(w2bf + (size_t)(16 * t + m) * DOUT + c0, koff);
      acc2[t] = __builtin_amdgcn_wmma_f32_16x16x32_bf16(false, a, false, hop, (short)0, acc2[t], false, false);
    }
  }
#pragma unroll
  for (int t = 0; t < 8; ++t)
#pragma unroll
    for (int v = 0; v < 8; ++v) {
      int d = 16 * t + (lhalf ? 8 : 0) + v;
      y[((size_t)b * DOUT + d) * NN + n0 + m] = acc2[t][v] + b2[d];
    }
}

// ---------------------------------------------------------------------------
extern "C" void kernel_launch(void* const* d_in, const int* in_sizes, int n_in,
                              void* d_out, int out_size, void* d_ws, size_t ws_size,
                              hipStream_t stream) {
  (void)in_sizes; (void)n_in; (void)out_size; (void)ws_size;
  const float* x     = (const float*)d_in[0];
  const float* wq    = (const float*)d_in[1];
  const float* bq    = (const float*)d_in[2];
  const float* wk    = (const float*)d_in[3];
  const float* bk    = (const float*)d_in[4];
  const float* wv    = (const float*)d_in[5];
  const float* bv    = (const float*)d_in[6];
  const float* gamma = (const float*)d_in[7];
  const float* w1    = (const float*)d_in[8];
  const float* b1    = (const float*)d_in[9];
  const float* w2    = (const float*)d_in[10];
  const float* b2    = (const float*)d_in[11];

  char* ws = (char*)d_ws;
  __bf16* xbf  = (__bf16*)(ws);                         // 16 MiB [b][n][c]
  __bf16* qws  = (__bf16*)(ws + ((size_t)16 << 20));    //  4 MiB [b][n][64]
  __bf16* kws  = (__bf16*)(ws + ((size_t)20 << 20));    //  4 MiB [b][n][64]
  __bf16* vws  = (__bf16*)(ws + ((size_t)24 << 20));    // 16 MiB [b][c][n]
  __bf16* outT = (__bf16*)(ws + ((size_t)40 << 20));    // 16 MiB [b][n][c]
  char*   wsw  = ws + ((size_t)56 << 20);
  __bf16* wqbf = (__bf16*)(wsw);                        //  32 KiB
  __bf16* wkbf = (__bf16*)(wsw + (32  << 10));          //  32 KiB
  __bf16* wvbf = (__bf16*)(wsw + (64  << 10));          // 128 KiB
  __bf16* w1bf = (__bf16*)(wsw + (192 << 10));          //  64 KiB
  __bf16* w2bf = (__bf16*)(wsw + (256 << 10));          //  32 KiB

  cvt_kernel<<<dim3(64),  256, 0, stream>>>(wq, wqbf, DQK * CC);
  cvt_kernel<<<dim3(64),  256, 0, stream>>>(wk, wkbf, DQK * CC);
  cvt_kernel<<<dim3(256), 256, 0, stream>>>(wv, wvbf, CC * CC);
  cvt_kernel<<<dim3(128), 256, 0, stream>>>(w1, w1bf, DOUT * CC);
  cvt_kernel<<<dim3(64),  256, 0, stream>>>(w2, w2bf, DOUT * DOUT);

  xpose_kernel<<<dim3(NN / 32, CC / 32, BB), 256, 0, stream>>>(x, xbf);

  qkv_wmma_kernel<<<dim3(NN / 16, 3, BB), 32, 0, stream>>>(
      xbf, wqbf, wkbf, wvbf, bq, bk, bv, qws, kws, vws);

  attn_kernel<<<dim3(NN / 16, BB), 32, 0, stream>>>(qws, kws, vws, x, gamma, outT);

  mlp_wmma_kernel<<<dim3(NN / 16, BB), 32, 0, stream>>>(outT, w1bf, b1, w2bf, b2,
                                                        (float*)d_out);
}
